// MultiHeadAttention_70712341562065
// MI455X (gfx1250) — compile-verified, matrix operand order checked by reading, not run on hardware
//
#include <hip/hip_runtime.h>
#include <hip/hip_bf16.h>

// ---------------------------------------------------------------------------
// MultiHeadAttention forward for MI455X (gfx1250, wave32, WMMA).
// GEMMs use v_wmma_f32_16x16x32_bf16 (f32 accumulate).
// scores+softmax+ctx fused: 16x2048 f32 score stripe lives in LDS (128KB of
// the 320KB/WGP), attn tensor is written to HBM exactly once.
// ---------------------------------------------------------------------------

typedef __attribute__((ext_vector_type(16))) __bf16 bf16x16;
typedef __attribute__((ext_vector_type(8)))  float  f32x8;

#define S_LEN  2048
#define DM     1024
#define NHEADS 16
#define DKH    64
#define BATCH  2

// round-to-nearest-even f32 -> bf16
__device__ __forceinline__ unsigned short f2bf(float x) {
  union { float f; unsigned int u; } c; c.f = x;
  unsigned int r = c.u + 0x7FFFu + ((c.u >> 16) & 1u);
  return (unsigned short)(r >> 16);
}

// 16x32 bf16 fragment from an LDS tile with row stride `stride` (ushorts).
// Per ISA 7.12.2: lanes 0-15 row M=lane, VGPR j holds K=(j<4?0:16)+(j&3)*2
// (+1); lanes 16-31 same rows, K offset +8.  (B uses the symmetric layout
// with tiles stored [n][k].)
__device__ __forceinline__ bf16x16 load_frag_s(const unsigned short* base, int stride) {
  const int lane = threadIdx.x & 31;
  const int hf   = lane >> 4;
  const int r    = lane & 15;
  union { unsigned int u[8]; bf16x16 v; } f;
  const unsigned int* row = (const unsigned int*)(base + r * stride);
#pragma unroll
  for (int j = 0; j < 8; ++j) {
    const int kbase = ((j & 4) << 2) + (hf << 3) + ((j & 3) << 1);
    f.u[j] = row[kbase >> 1];
  }
  return f.v;
}

// A-fragment built from an f32 LDS tile (row stride in floats), cvt -> bf16.
__device__ __forceinline__ bf16x16 frag_from_f32(const float* base, int stride) {
  const int lane = threadIdx.x & 31;
  const int hf   = lane >> 4;
  const int r    = lane & 15;
  union { unsigned int u[8]; bf16x16 v; } f;
  const float* row = base + r * stride;
#pragma unroll
  for (int j = 0; j < 8; ++j) {
    const int kbase = ((j & 4) << 2) + (hf << 3) + ((j & 3) << 1);
    float2 p = *(const float2*)(row + kbase);
    f.u[j] = ((unsigned int)f2bf(p.y) << 16) | (unsigned int)f2bf(p.x);
  }
  return f.v;
}

// One 32-deep K step for an 8-wave 128x64 block tile (4x2 wave grid,
// each wave owns a 32x32 tile = 2x2 WMMAs).
__device__ __forceinline__ void wmma_step(const unsigned short* As,
                                          const unsigned short* Bs,
                                          int wm, int wn, f32x8 c[2][2]) {
  bf16x16 a0 = load_frag_s(As + (wm * 32 + 0)  * 32, 32);
  bf16x16 a1 = load_frag_s(As + (wm * 32 + 16) * 32, 32);
  bf16x16 b0 = load_frag_s(Bs + (wn * 32 + 0)  * 32, 32);
  bf16x16 b1 = load_frag_s(Bs + (wn * 32 + 16) * 32, 32);
  c[0][0] = __builtin_amdgcn_wmma_f32_16x16x32_bf16(false, a0, false, b0, (short)0, c[0][0], false, false);
  c[0][1] = __builtin_amdgcn_wmma_f32_16x16x32_bf16(false, a0, false, b1, (short)0, c[0][1], false, false);
  c[1][0] = __builtin_amdgcn_wmma_f32_16x16x32_bf16(false, a1, false, b0, (short)0, c[1][0], false, false);
  c[1][1] = __builtin_amdgcn_wmma_f32_16x16x32_bf16(false, a1, false, b1, (short)0, c[1][1], false, false);
}

// ---------------------------------------------------------------------------
// Kernel 1: projection  Out_bf16[b][h][s][dk] = (A_f32 @ W_f32 + bias)
// ---------------------------------------------------------------------------
__global__ __launch_bounds__(256) void proj_kernel(
    const float* __restrict__ A, const float* __restrict__ W,
    const float* __restrict__ bias, unsigned short* __restrict__ Out) {
  __shared__ __align__(16) unsigned short As[128 * 32];
  __shared__ __align__(16) unsigned short Bs[64 * 32];
  const int m0 = blockIdx.x * 128;
  const int n0 = blockIdx.y * 64;
  const int tid = threadIdx.x, lane = tid & 31, wave = tid >> 5;
  const int wm = wave >> 1, wn = wave & 1;
  f32x8 c[2][2] = {};

  for (int k0 = 0; k0 < DM; k0 += 32) {
    __syncthreads();
    {  // A panel: 128x32, f32 -> bf16
      const int r = tid >> 1, c4 = (tid & 1) * 16;
      const float* src = A + (size_t)(m0 + r) * DM + k0 + c4;
      unsigned short* dst = As + r * 32 + c4;
#pragma unroll
      for (int i = 0; i < 16; i += 4) {
        float4 f = *(const float4*)(src + i);
        dst[i]     = f2bf(f.x); dst[i + 1] = f2bf(f.y);
        dst[i + 2] = f2bf(f.z); dst[i + 3] = f2bf(f.w);
      }
    }
    {  // B panel: W transposed into Bs[n][k], f32 -> bf16
      const int kk = tid >> 3, nb = (tid & 7) * 8;
      const float* src = W + (size_t)(k0 + kk) * DM + n0 + nb;
#pragma unroll
      for (int i = 0; i < 8; i += 4) {
        float4 f = *(const float4*)(src + i);
        Bs[(nb + i)     * 32 + kk] = f2bf(f.x);
        Bs[(nb + i + 1) * 32 + kk] = f2bf(f.y);
        Bs[(nb + i + 2) * 32 + kk] = f2bf(f.z);
        Bs[(nb + i + 3) * 32 + kk] = f2bf(f.w);
      }
    }
    __syncthreads();
    wmma_step(As, Bs, wm, wn, c);
  }

  const int hf = lane >> 4, nl = lane & 15;
#pragma unroll
  for (int tm = 0; tm < 2; ++tm)
#pragma unroll
    for (int tn = 0; tn < 2; ++tn)
#pragma unroll
      for (int v = 0; v < 8; ++v) {
        const int m = m0 + wm * 32 + tm * 16 + v + 8 * hf;
        const int n = n0 + wn * 32 + tn * 16 + nl;
        const float val = c[tm][tn][v] + bias[n];
        const int b = m >> 11, s = m & (S_LEN - 1);
        const int h = n >> 6,  d = n & (DKH - 1);
        Out[(((size_t)b * NHEADS + h) * S_LEN + s) * DKH + d] = f2bf(val);
      }
}

// ---------------------------------------------------------------------------
// Kernel 2 (fused): per (bh, 16-row q-stripe):
//   scores (WMMA) -> LDS f32 stripe -> softmax in LDS -> attn written once
//   -> ctx = P @ V (WMMA, causal K-truncation) -> Ctx bf16 (model layout).
// Dynamic LDS layout (bytes):
//   Srow  [16][2048] f32   131072
//   Kt    [128][64] bf16    16384   (keys x dk)
//   Qt    [16][64]  bf16     2048
//   Vt    [64][64]  bf16     8192   ([d][k] transposed)
//   red   [16][16]  f32      1024
//   Cpart [4][16][16] f32    4096
// ---------------------------------------------------------------------------
#define FUSED_SMEM (131072 + 16384 + 2048 + 8192 + 1024 + 4096)

__global__ __launch_bounds__(256) void fused_attn_kernel(
    const unsigned short* __restrict__ Q, const unsigned short* __restrict__ K,
    const unsigned short* __restrict__ V, float* __restrict__ attn,
    unsigned short* __restrict__ Ctx) {
  extern __shared__ char smem[];
  float*          Srow  = (float*)smem;                          // [16][2048]
  unsigned short* Kt    = (unsigned short*)(smem + 131072);      // [128][64]
  unsigned short* Qt    = Kt + 128 * 64;                         // [16][64]
  unsigned short* Vt    = Qt + 16 * 64;                          // [64][64]
  float*          red   = (float*)(Vt + 64 * 64);                // [16][16]
  float*          Cpart = red + 256;                             // [4][16][16]

  const int bh = blockIdx.y;
  const int b  = bh >> 4, h = bh & (NHEADS - 1);
  const int q0 = blockIdx.x * 16;
  const int tid = threadIdx.x, lane = tid & 31, wave = tid >> 5;
  const int hf = lane >> 4, nl = lane & 15;

  const unsigned short* Qb = Q + (size_t)bh * S_LEN * DKH;
  const unsigned short* Kb = K + (size_t)bh * S_LEN * DKH;
  const unsigned short* Vb = V + (size_t)bh * S_LEN * DKH;
  float* Ab = attn + (size_t)bh * S_LEN * S_LEN;

  const int ncol   = q0 + 16;                 // valid key prefix (causal)
  const int ncol64 = (ncol + 63) & ~63;       // padded to WMMA K granularity

  // ---- stage Q stripe, keep A fragments in registers for the whole kernel
  if (tid < 128) {
    const int r = tid >> 3, c8 = (tid & 7) * 8;
    *(uint4*)(Qt + r * 64 + c8) = *(const uint4*)(Qb + (size_t)(q0 + r) * DKH + c8);
  }
  __syncthreads();
  const bf16x16 a0 = load_frag_s(Qt, 64);        // dk 0..31
  const bf16x16 a1 = load_frag_s(Qt + 32, 64);   // dk 32..63

  // ---- score phase: 128-key superblocks, each wave owns one 16-key tile
  for (int sb = 0; sb < ncol; sb += 128) {
    __syncthreads();
    {  // stage Kt[128][64]
      const int r = tid >> 1, c8 = (tid & 1) * 32;
      const unsigned short* src = Kb + (size_t)(sb + r) * DKH + c8;
      uint4* dst = (uint4*)(Kt + r * 64 + c8);
      dst[0] = ((const uint4*)src)[0]; dst[1] = ((const uint4*)src)[1];
      dst[2] = ((const uint4*)src)[2]; dst[3] = ((const uint4*)src)[3];
      // prefetch next superblock's K rows (global_prefetch_b8)
      if (sb + 128 < ncol && (sb + 128 + r) < S_LEN)
        __builtin_prefetch(src + (size_t)128 * DKH, 0, 1);
    }
    __syncthreads();
    const int t0 = sb + wave * 16;
    if (t0 < ncol) {  // wave-uniform branch: EXEC stays full for WMMA
      bf16x16 b0 = load_frag_s(Kt + (wave * 16) * 64, 64);
      bf16x16 b1 = load_frag_s(Kt + (wave * 16) * 64 + 32, 64);
      f32x8 c = {};
      c = __builtin_amdgcn_wmma_f32_16x16x32_bf16(false, a0, false, b0, (short)0, c, false, false);
      c = __builtin_amdgcn_wmma_f32_16x16x32_bf16(false, a1, false, b1, (short)0, c, false, false);
#pragma unroll
      for (int v = 0; v < 8; ++v) {
        const int m   = v + 8 * hf;
        const int col = t0 + nl;
        float val = c[v] * 0.125f;           // 1/sqrt(64)
        if (col > q0 + m) val = -1e9f;       // causal mask
        Srow[m * S_LEN + col] = val;
      }
    }
  }
  __syncthreads();

  // ---- softmax over [0, ncol): 16 threads per row
  {
    const int r = tid >> 4, li = tid & 15;
    float* row = Srow + r * S_LEN;
    float mx = -3.0e38f;
    for (int c2 = li; c2 < ncol; c2 += 16) mx = fmaxf(mx, row[c2]);
    red[r * 16 + li] = mx;
    __syncthreads();
    if (li == 0) {
      float m2 = red[r * 16];
      for (int i = 1; i < 16; ++i) m2 = fmaxf(m2, red[r * 16 + i]);
      red[r * 16] = m2;
    }
    __syncthreads();
    mx = red[r * 16];
    __syncthreads();
    float sum = 0.f;
    for (int c2 = li; c2 < ncol; c2 += 16) {
      const float e = __expf(row[c2] - mx);
      row[c2] = e; sum += e;
    }
    red[r * 16 + li] = sum;
    __syncthreads();
    if (li == 0) {
      float s2 = 0.f;
      for (int i = 0; i < 16; ++i) s2 += red[r * 16 + i];
      red[r * 16] = s2;
    }
    __syncthreads();
    const float inv = 1.f / red[r * 16];
    for (int c2 = li; c2 < ncol; c2 += 16) row[c2] *= inv;
  }
  __syncthreads();

  // ---- write attn (once), zero the masked suffix, zero-pad stripe to ncol64
#pragma unroll 1
  for (int r = 0; r < 16; ++r) {
    float* dst = Ab + (size_t)(q0 + r) * S_LEN;
    const float* src = Srow + r * S_LEN;
    for (int c2 = tid; c2 < ncol; c2 += 256)          dst[c2] = src[c2];
    for (int c2 = ncol + tid; c2 < S_LEN; c2 += 256)  dst[c2] = 0.f;
  }
#pragma unroll 1
  for (int r = 0; r < 16; ++r)
    for (int c2 = ncol + tid; c2 < ncol64; c2 += 256) Srow[r * S_LEN + c2] = 0.f;

  // ---- ctx = P @ V: 4 d-tiles x 2 K-phases across 8 waves
  const int d0   = (wave & 3) * 16;
  const int koff = (wave >> 2) * 32;
  f32x8 cc = {};
  for (int kb = 0; kb < ncol64; kb += 64) {
    __syncthreads();
    {  // stage Vt[d][k] for keys kb..kb+63 (transpose)
      const int kk = tid >> 2, db = (tid & 3) * 16;
      const unsigned short* src = Vb + (size_t)(kb + kk) * DKH + db;
#pragma unroll
      for (int i = 0; i < 16; ++i) Vt[(db + i) * 64 + kk] = src[i];
      // prefetch next V tile (global_prefetch_b8)
      if (kb + 64 < ncol64 && (kb + 64 + kk) < S_LEN)
        __builtin_prefetch(src + (size_t)64 * DKH, 0, 1);
    }
    __syncthreads();
    bf16x16 a = frag_from_f32(Srow + kb + koff, S_LEN);
    bf16x16 bb = load_frag_s(Vt + d0 * 64 + koff, 64);
    cc = __builtin_amdgcn_wmma_f32_16x16x32_bf16(false, a, false, bb, (short)0, cc, false, false);
  }
  __syncthreads();
  if (wave >= 4) {
#pragma unroll
    for (int v = 0; v < 8; ++v)
      Cpart[(wave - 4) * 256 + (v + 8 * hf) * 16 + nl] = cc[v];
  }
  __syncthreads();
  if (wave < 4) {
#pragma unroll
    for (int v = 0; v < 8; ++v) {
      const int m = v + 8 * hf;
      const float val = cc[v] + Cpart[wave * 256 + m * 16 + nl];
      const int s = q0 + m;
      Ctx[((size_t)(b * S_LEN + s)) * DM + h * DKH + d0 + nl] = f2bf(val);
    }
  }
}

// ---------------------------------------------------------------------------
// Kernel 3: out = Ctx_bf16 @ Wo_f32 + bo, f32 output.
// ---------------------------------------------------------------------------
__global__ __launch_bounds__(256) void outproj_kernel(
    const unsigned short* __restrict__ A, const float* __restrict__ W,
    const float* __restrict__ bias, float* __restrict__ out) {
  __shared__ __align__(16) unsigned short As[128 * 32];
  __shared__ __align__(16) unsigned short Bs[64 * 32];
  const int m0 = blockIdx.x * 128;
  const int n0 = blockIdx.y * 64;
  const int tid = threadIdx.x, lane = tid & 31, wave = tid >> 5;
  const int wm = wave >> 1, wn = wave & 1;
  f32x8 c[2][2] = {};

  for (int k0 = 0; k0 < DM; k0 += 32) {
    __syncthreads();
    {  // A: bf16 copy
      const int r = tid >> 1, c4 = (tid & 1) * 16;
      const unsigned short* src = A + (size_t)(m0 + r) * DM + k0 + c4;
      uint4* dst = (uint4*)(As + r * 32 + c4);
      dst[0] = ((const uint4*)src)[0];
      dst[1] = ((const uint4*)src)[1];
    }
    {  // B: Wo transpose + convert
      const int kk = tid >> 3, nb = (tid & 7) * 8;
      const float* src = W + (size_t)(k0 + kk) * DM + n0 + nb;
#pragma unroll
      for (int i = 0; i < 8; i += 4) {
        float4 f = *(const float4*)(src + i);
        Bs[(nb + i)     * 32 + kk] = f2bf(f.x);
        Bs[(nb + i + 1) * 32 + kk] = f2bf(f.y);
        Bs[(nb + i + 2) * 32 + kk] = f2bf(f.z);
        Bs[(nb + i + 3) * 32 + kk] = f2bf(f.w);
      }
    }
    __syncthreads();
    wmma_step(As, Bs, wm, wn, c);
  }

  const int hf = lane >> 4, nl = lane & 15;
#pragma unroll
  for (int tm = 0; tm < 2; ++tm)
#pragma unroll
    for (int tn = 0; tn < 2; ++tn)
#pragma unroll
      for (int v = 0; v < 8; ++v) {
        const int m = m0 + wm * 32 + tm * 16 + v + 8 * hf;
        const int n = n0 + wn * 32 + tn * 16 + nl;
        out[(size_t)m * DM + n] = c[tm][tn][v] + bias[n];
      }
}

// ---------------------------------------------------------------------------
extern "C" void kernel_launch(void* const* d_in, const int* in_sizes, int n_in,
                              void* d_out, int out_size, void* d_ws, size_t ws_size,
                              hipStream_t stream) {
  const float* q  = (const float*)d_in[0];
  const float* k  = (const float*)d_in[1];
  const float* v  = (const float*)d_in[2];
  // d_in[3] = causal mask (implemented analytically)
  const float* wq = (const float*)d_in[4];
  const float* bq = (const float*)d_in[5];
  const float* wk = (const float*)d_in[6];
  const float* bk = (const float*)d_in[7];
  const float* wv = (const float*)d_in[8];
  const float* bv = (const float*)d_in[9];
  const float* wo = (const float*)d_in[10];
  const float* bo = (const float*)d_in[11];

  float* out  = (float*)d_out;                               // [B, S, DM]
  float* attn = (float*)d_out + (size_t)BATCH * S_LEN * DM;  // [B, H, S, S]

  // Workspace (bf16 as ushort): Qh | Kh | Vh (head-major) | Ctx (model layout)
  const size_t HSZ = (size_t)BATCH * NHEADS * S_LEN * DKH;   // 4,194,304
  unsigned short* Qh  = (unsigned short*)d_ws;
  unsigned short* Kh  = Qh + HSZ;
  unsigned short* Vh  = Kh + HSZ;
  unsigned short* Ctx = Vh + HSZ;

  // Unconditional (idempotent, deterministic, capture-safe): raise the
  // dynamic-LDS cap for the fused kernel.
  (void)hipFuncSetAttribute((const void*)fused_attn_kernel,
                            hipFuncAttributeMaxDynamicSharedMemorySize,
                            FUSED_SMEM);

  const dim3 blk(256);
  proj_kernel<<<dim3(32, 16), blk, 0, stream>>>(q, wq, bq, Qh);
  proj_kernel<<<dim3(32, 16), blk, 0, stream>>>(k, wk, bk, Kh);
  proj_kernel<<<dim3(32, 16), blk, 0, stream>>>(v, wv, bv, Vh);
  fused_attn_kernel<<<dim3(S_LEN / 16, BATCH * NHEADS), blk, FUSED_SMEM, stream>>>(
      Qh, Kh, Vh, attn, Ctx);
  outproj_kernel<<<dim3(32, 16), blk, 0, stream>>>(Ctx, wo, bo, out);
}
